// GAT2_72653666779146
// MI455X (gfx1250) — compile-verified, matrix operand order checked by reading, not run on hardware
//
#include <hip/hip_runtime.h>
#include <cmath>
#include <cstdint>

typedef __attribute__((ext_vector_type(2)))  float  v2f;
typedef __attribute__((ext_vector_type(8)))  float  v8f;
typedef __attribute__((ext_vector_type(16))) __bf16 v16bf;

#if defined(__HIP_DEVICE_COMPILE__) && __has_builtin(__builtin_amdgcn_wmma_f32_16x16x4_f32)
#define USE_WMMA_F32 1
#else
#define USE_WMMA_F32 0
#endif

__device__ __forceinline__ void async_wait0() {
#if defined(__HIP_DEVICE_COMPILE__) && __has_builtin(__builtin_amdgcn_s_wait_asynccnt)
  __builtin_amdgcn_s_wait_asynccnt(0);
#else
  asm volatile("s_wait_asynccnt 0x0" ::: "memory");
#endif
}

// Async copy of one dword global -> LDS (CDNA5 GLOBAL_LOAD_ASYNC_TO_LDS_B32,
// tracked by ASYNCcnt). ldsOff is the 32-bit LDS offset (low 32 bits of the
// generic pointer per ISA aperture mapping).
__device__ __forceinline__ void async_copy_b32(unsigned ldsOff, const void* gaddr) {
  asm volatile("global_load_async_to_lds_b32 %0, %1, off"
               :: "v"(ldsOff), "v"(gaddr) : "memory");
}

// ---------------------------------------------------------------------------
// GEMM: C[M,256] = A[M,K] * W[K,256], fp32, one wave per 16x16 output tile.
// ---------------------------------------------------------------------------
__global__ __launch_bounds__(128)
void gat_gemm_wmma(const float* __restrict__ A, const float* __restrict__ W,
                   float* __restrict__ C, int M, int K) {
  const int NCOL = 256;
  int lane  = threadIdx.x & 31;
  int wave  = threadIdx.x >> 5;
  int tileM = blockIdx.x;                 // M/16 tiles
  int tileN = blockIdx.y * 4 + wave;      // 16 column tiles of 16
  int m     = tileM * 16 + (lane & 15);
  int col   = tileN * 16 + (lane & 15);
  int hi    = (lane >> 4);                // lane half

  v8f acc = {0.f, 0.f, 0.f, 0.f, 0.f, 0.f, 0.f, 0.f};

#if USE_WMMA_F32
  int khalf = hi * 2;
  for (int k = 0; k < K; k += 4) {
    v2f a, b;
    a.x = A[(size_t)m * K + k + khalf + 0];
    a.y = A[(size_t)m * K + k + khalf + 1];
    b.x = W[(size_t)(k + khalf + 0) * NCOL + col];
    b.y = W[(size_t)(k + khalf + 1) * NCOL + col];
    acc = __builtin_amdgcn_wmma_f32_16x16x4_f32(
        /*neg_a=*/false, a, /*neg_b=*/false, b,
        /*c_mod=*/(short)0, acc, /*reuse_a=*/false, /*reuse_b=*/false);
  }
#else
  // bf16 fallback (probe-confirmed builtin).
  for (int k = 0; k < K; k += 32) {
    int kb = k + (hi ? 8 : 0);
    v16bf a, b;
#pragma unroll
    for (int i = 0; i < 8; ++i) {
      a[i]     = (__bf16)A[(size_t)m * K + kb + i];
      a[8 + i] = (__bf16)A[(size_t)m * K + kb + 16 + i];
      b[i]     = (__bf16)W[(size_t)(kb + i) * NCOL + col];
      b[8 + i] = (__bf16)W[(size_t)(kb + 16 + i) * NCOL + col];
    }
    acc = __builtin_amdgcn_wmma_f32_16x16x32_bf16(
        false, a, false, b, (short)0, acc, false, false);
  }
#endif

  int rbase = tileM * 16 + (hi ? 8 : 0);
#pragma unroll
  for (int r = 0; r < 8; ++r)
    C[(size_t)(rbase + r) * NCOL + col] = acc[r];
}

// ---------------------------------------------------------------------------
// Attention projections: el[n,h] = sum_d h[n,h*D+d]*al[h*D+d] (same for er).
// ---------------------------------------------------------------------------
__global__ __launch_bounds__(256)
void gat_attn(const float* __restrict__ h, const float* __restrict__ al,
              const float* __restrict__ ar, float* __restrict__ el,
              float* __restrict__ er, int D) {
  __shared__ float sl[256], sr[256];
  int n = blockIdx.x, c = threadIdx.x;
  float v = h[(size_t)n * 256 + c];
  sl[c] = v * al[c];
  sr[c] = v * ar[c];
  __syncthreads();
  for (int s = D >> 1; s > 0; s >>= 1) {
    if ((c & (D - 1)) < s) { sl[c] += sl[c + s]; sr[c] += sr[c + s]; }
    __syncthreads();
  }
  if ((c & (D - 1)) == 0) {
    int hh = c / D;
    int H = 256 / D;
    el[(size_t)n * H + hh] = sl[c];
    er[(size_t)n * H + hh] = sr[c];
  }
}

// ---------------------------------------------------------------------------
// CSR-by-dst construction: count, exclusive scan, fill.
// ---------------------------------------------------------------------------
__global__ void gat_count(const int* __restrict__ dst, int* __restrict__ cnt, int E) {
  int e = blockIdx.x * blockDim.x + threadIdx.x;
  if (e < E) atomicAdd(&cnt[dst[e]], 1);
}

__global__ __launch_bounds__(1024)
void gat_scan(const int* __restrict__ cnt, int* __restrict__ rowp,
              int* __restrict__ cur, int N) {
  __shared__ int lds[1024];
  int tid = threadIdx.x;
  int carry = 0;
  for (int base = 0; base < N; base += 1024) {
    int v = (base + tid < N) ? cnt[base + tid] : 0;
    lds[tid] = v;
    __syncthreads();
    for (int off = 1; off < 1024; off <<= 1) {
      int t = (tid >= off) ? lds[tid - off] : 0;
      __syncthreads();
      lds[tid] += t;
      __syncthreads();
    }
    int excl = lds[tid] - v;
    if (base + tid < N) { rowp[base + tid] = carry + excl; cur[base + tid] = carry + excl; }
    int total = lds[1023];
    __syncthreads();
    carry += total;
  }
  if (tid == 0) rowp[N] = carry;
}

__global__ void gat_fill(const int* __restrict__ dst, int* __restrict__ cur,
                         int* __restrict__ perm, int E) {
  int e = blockIdx.x * blockDim.x + threadIdx.x;
  if (e < E) { int p = atomicAdd(&cur[dst[e]], 1); perm[p] = e; }
}

// ---------------------------------------------------------------------------
// Per-destination-node edge softmax + aggregation + bias + ELU.
// One 256-thread workgroup per node; zero float atomics.
// ---------------------------------------------------------------------------
__global__ __launch_bounds__(256)
void gat_aggregate(const float* __restrict__ hsrc, const int* __restrict__ src,
                   const int* __restrict__ rowp, const int* __restrict__ perm,
                   const float* __restrict__ el, const float* __restrict__ er,
                   const float* __restrict__ bias, float* __restrict__ out,
                   int H, int D) {
  __shared__ int   eids[1024];
  __shared__ float red[256];
  __shared__ float mh[4], sh[4];
  int n = blockIdx.x, tid = threadIdx.x;
  int start = rowp[n];
  int deg   = rowp[n + 1] - start;
  bool inLds = (deg <= 1024);

  // Stage edge ids into LDS with CDNA5 async global->LDS copies, then sort
  // by edge id (deterministic fp accumulation order across replays).
  if (inLds) {
    for (int j = tid; j < deg; j += 256) {
      async_copy_b32((unsigned)(uintptr_t)&eids[j], perm + start + j);
    }
    async_wait0();
    __syncthreads();
    for (int r = 0; r < deg; ++r) {
      int par = r & 1;
      for (int j = tid;; j += 256) {
        int i0 = 2 * j + par;
        if (i0 + 1 >= deg) break;
        int a = eids[i0], b = eids[i0 + 1];
        if (a > b) { eids[i0] = b; eids[i0 + 1] = a; }
      }
      __syncthreads();
    }
  }

  float er_n[4];
  for (int h = 0; h < H; ++h) er_n[h] = er[(size_t)n * H + h];

  // Pass 1: per-head max of leaky_relu(el[src]+er[n]).
  float lm[4] = {-__builtin_inff(), -__builtin_inff(), -__builtin_inff(), -__builtin_inff()};
  for (int j = tid; j < deg; j += 256) {
    int e = inLds ? eids[j] : perm[start + j];
    int s = src[e];
    for (int h = 0; h < H; ++h) {
      float x = el[(size_t)s * H + h] + er_n[h];
      float sc = (x > 0.f) ? x : 0.2f * x;
      lm[h] = fmaxf(lm[h], sc);
    }
  }
  for (int h = 0; h < H; ++h) {
    red[tid] = lm[h];
    __syncthreads();
    for (int s2 = 128; s2 > 0; s2 >>= 1) {
      if (tid < s2) red[tid] = fmaxf(red[tid], red[tid + s2]);
      __syncthreads();
    }
    if (tid == 0) mh[h] = red[0];
    __syncthreads();
  }

  // Pass 2: per-head sum of exp(score - max).
  float ls[4] = {0.f, 0.f, 0.f, 0.f};
  for (int j = tid; j < deg; j += 256) {
    int e = inLds ? eids[j] : perm[start + j];
    int s = src[e];
    for (int h = 0; h < H; ++h) {
      float x = el[(size_t)s * H + h] + er_n[h];
      float sc = (x > 0.f) ? x : 0.2f * x;
      ls[h] += expf(sc - mh[h]);
    }
  }
  for (int h = 0; h < H; ++h) {
    red[tid] = ls[h];
    __syncthreads();
    for (int s2 = 128; s2 > 0; s2 >>= 1) {
      if (tid < s2) red[tid] += red[tid + s2];
      __syncthreads();
    }
    if (tid == 0) sh[h] = red[0];
    __syncthreads();
  }

  // Pass 3: weighted accumulation of source rows (coalesced 1 KB row reads).
  int c = tid;
  int h = c / D;
  float rm  = mh[h];
  float rs  = fmaxf(sh[h], 1e-9f);
  float ern = er_n[h];
  float acc = 0.f;
  for (int j = 0; j < deg; ++j) {
    int e = inLds ? eids[j] : perm[start + j];
    int s = src[e];
    if (j + 1 < deg) {
      int e2 = inLds ? eids[j + 1] : perm[start + j + 1];
      __builtin_prefetch(&hsrc[(size_t)src[e2] * 256 + c], 0, 0);  // global_prefetch_b8
    }
    float x  = el[(size_t)s * H + h] + ern;
    float sc = (x > 0.f) ? x : 0.2f * x;
    float alpha = expf(sc - rm) / rs;
    acc += alpha * hsrc[(size_t)s * 256 + c];
  }
  float o = acc + bias[c];
  out[(size_t)n * 256 + c] = (o > 0.f) ? o : expm1f(o);   // ELU
}

// ---------------------------------------------------------------------------
// Final gather: out[i,:] = hf[x[i],:], float4-vectorized (b128 ld/st).
// ---------------------------------------------------------------------------
__global__ __launch_bounds__(64)
void gat_gather(const float* __restrict__ hf, const int* __restrict__ x,
                float* __restrict__ out) {
  int i = blockIdx.x;
  int idx = x[i];
  const float4* s = (const float4*)(hf + (size_t)idx * 256);
  float4*       d = (float4*)(out + (size_t)i * 256);
  d[threadIdx.x] = s[threadIdx.x];
}

// ---------------------------------------------------------------------------
extern "C" void kernel_launch(void* const* d_in, const int* in_sizes, int n_in,
                              void* d_out, int out_size, void* d_ws, size_t ws_size,
                              hipStream_t stream) {
  (void)in_sizes; (void)n_in; (void)out_size; (void)ws_size;
  const float* feat = (const float*)d_in[0];
  const int*   src  = (const int*)d_in[1];
  const int*   dst  = (const int*)d_in[2];
  const int*   x    = (const int*)d_in[3];
  const float* W1   = (const float*)d_in[4];
  const float* al1  = (const float*)d_in[5];
  const float* ar1  = (const float*)d_in[6];
  const float* b1   = (const float*)d_in[7];
  const float* W2   = (const float*)d_in[8];
  const float* al2  = (const float*)d_in[9];
  const float* ar2  = (const float*)d_in[10];
  const float* b2   = (const float*)d_in[11];
  float* out = (float*)d_out;

  const int N = 50000, E = 800000, Fin = 128, F = 256;

  // Workspace layout (~108 MB):
  float* hA  = (float*)d_ws;                 // [N,256] h_pre (layer1 then layer2)
  float* hB  = hA + (size_t)N * F;           // [N,256] activated output
  float* el  = hB + (size_t)N * F;           // [N,4]
  float* er  = el + (size_t)N * 4;           // [N,4]
  int* cnt   = (int*)(er + (size_t)N * 4);   // [N]
  int* rowp  = cnt + N;                      // [N+1]
  int* cur   = rowp + (N + 1);               // [N]
  int* perm  = cur + N;                      // [E]

  // CSR by destination (shared by both layers).
  hipMemsetAsync(cnt, 0, (size_t)N * sizeof(int), stream);
  gat_count<<<(E + 255) / 256, 256, 0, stream>>>(dst, cnt, E);
  gat_scan<<<1, 1024, 0, stream>>>(cnt, rowp, cur, N);
  gat_fill<<<(E + 255) / 256, 256, 0, stream>>>(dst, cur, perm, E);

  dim3 ggrid(N / 16, 4);

  // Layer 1: H=4, D=64
  gat_gemm_wmma<<<ggrid, 128, 0, stream>>>(feat, W1, hA, N, Fin);
  gat_attn<<<N, 256, 0, stream>>>(hA, al1, ar1, el, er, 64);
  gat_aggregate<<<N, 256, 0, stream>>>(hA, src, rowp, perm, el, er, b1, hB, 4, 64);

  // Layer 2: H=1, D=256
  gat_gemm_wmma<<<ggrid, 128, 0, stream>>>(hB, W2, hA, N, F);
  gat_attn<<<N, 256, 0, stream>>>(hA, al2, ar2, el, er, 256);
  gat_aggregate<<<N, 256, 0, stream>>>(hA, src, rowp, perm, el, er, b2, hB, 1, 256);

  // Gather [16,2048,256]
  gat_gather<<<16 * 2048, 64, 0, stream>>>(hB, x, out);
}
